// ConvLstm_54400055771926
// MI455X (gfx1250) — compile-verified, matrix operand order checked by reading, not run on hardware
//
#include <hip/hip_runtime.h>
#include <math.h>

// ---------------- types ----------------
typedef __bf16 bf16_t;
typedef __attribute__((ext_vector_type(16))) __bf16 v16bf;
typedef __attribute__((ext_vector_type(8)))  __bf16 v8bf;
typedef __attribute__((ext_vector_type(8)))  float  v8f;

// ---------------- problem constants ----------------
#define BATCH 64
#define SEQ   256
#define HID   1024
#define GATES 4
#define K2    2048            // K = H (x side) + H (h side)
#define NTILE 64              // H columns per workgroup (per gate)
#define PAD   8               // bf16 elements of row padding (16B -> bank stagger)
#define LDA   (K2 + PAD)      // LDS row stride in bf16 elements (2056)

static __device__ __forceinline__ v16bf concat8(v8bf lo, v8bf hi) {
    return __builtin_shufflevector(lo, hi, 0,1,2,3,4,5,6,7,8,9,10,11,12,13,14,15);
}
static __device__ __forceinline__ float sigmoidf_(float x) {
    return 1.0f / (1.0f + __expf(-x));
}

// CDNA5 async global->LDS copy, 16B per lane, tracked by ASYNCcnt.
// (inline asm per bridge doc: portable across ROCm7.2 / amdgpu-toolchain)
static __device__ __forceinline__ void async_copy_b128(unsigned lds_byte_off,
                                                       unsigned long long gaddr) {
    asm volatile("global_load_async_to_lds_b128 %0, %1, off"
                 :: "v"(lds_byte_off), "v"(gaddr) : "memory");
}
static __device__ __forceinline__ void wait_async0() {
    asm volatile("s_wait_asynccnt 0x0" ::: "memory");
}

// ---------------- weight packing: Wcat[g][k][0:1024]=Wx, [1024:2048]=Wh, bf16 ----------------
__global__ void convlstm_pack_weights(const float* __restrict__ Wx,
                                      const float* __restrict__ Wh,
                                      bf16_t* __restrict__ Wcat) {
    size_t idx = (size_t)blockIdx.x * blockDim.x + threadIdx.x;   // over 4*1024*2048
    const size_t total = (size_t)GATES * HID * K2;
    if (idx >= total) return;
    size_t hh  = idx & (K2 - 1);
    size_t row = idx >> 11;                                       // g*HID + k
    float v = (hh < HID) ? Wx[row * HID + hh] : Wh[row * HID + (hh - HID)];
    Wcat[idx] = (bf16_t)v;
}

// ---------------- one-shot x fp32 -> bf16 (removes per-step conversion work) ----------------
__global__ void convlstm_convert_x(const float* __restrict__ x,
                                   bf16_t* __restrict__ xbf) {
    size_t i = ((size_t)blockIdx.x * blockDim.x + threadIdx.x) * 8;   // total = B*S*H
    float4 a = *(const float4*)(x + i);
    float4 b = *(const float4*)(x + i + 4);
    v8bf v = { (bf16_t)a.x, (bf16_t)a.y, (bf16_t)a.z, (bf16_t)a.w,
               (bf16_t)b.x, (bf16_t)b.y, (bf16_t)b.z, (bf16_t)b.w };
    *(v8bf*)(xbf + i) = v;
}

// ---------------- combined bias: bx + bh + bgate ----------------
__global__ void convlstm_pack_bias(const float* __restrict__ bx,
                                   const float* __restrict__ bh,
                                   const float* __restrict__ bg,
                                   float* __restrict__ bias) {
    int i = blockIdx.x * blockDim.x + threadIdx.x;
    if (i < GATES * HID) bias[i] = bx[i] + bh[i] + bg[i];
}

// ---------------- state init: h in bf16 (GEMM operand), c in fp32 ----------------
__global__ void convlstm_init_state(const float* __restrict__ h0,
                                    const float* __restrict__ c0,
                                    bf16_t* __restrict__ hbf,
                                    float* __restrict__ cbuf) {
    int i = blockIdx.x * blockDim.x + threadIdx.x;
    if (i < BATCH * HID) { hbf[i] = (bf16_t)h0[i]; cbuf[i] = c0[i]; }
}

// ---------------- one recurrence step ----------------
// grid.x = HID/NTILE = 16 workgroups; block = 256 threads = 8 waves (wave32).
// Each workgroup: all 4 gates for H-slice [n0, n0+64), all 64 batch rows.
// LDS: A panel [64][LDA] bf16 (257KB, CDNA5 320KB WGP LDS); overlaid with
//      gate buffer [4][64][64] fp32 (64KB) for the fused cell-update epilogue.
// A panel is filled with global_load_async_to_lds_b128 (ASYNCcnt path).
__launch_bounds__(256, 1)
__global__ void convlstm_step(const bf16_t* __restrict__ xbf,
                              const bf16_t* __restrict__ Wcat,
                              const float* __restrict__ bias,
                              const float* __restrict__ peep,
                              const bf16_t* __restrict__ h_in,
                              bf16_t* __restrict__ h_out,
                              float* __restrict__ cbuf,
                              float* __restrict__ out,
                              int s) {
    extern __shared__ char smem[];
    bf16_t* A  = (bf16_t*)smem;          // [BATCH][LDA] bf16
    float*  Gb = (float*)smem;           // overlaid [GATES][64][64] fp32 (epilogue)

    const int tid = threadIdx.x;
    const int n0  = blockIdx.x * NTILE;

    // ---- async-stage A = [xbf[:,s,:] | h_in] into LDS (pure byte copy, no VALU cvt) ----
    {
        const int row = tid >> 2;            // 0..63 (batch)
        const int seg = tid & 3;             // 4 x 512-element segments per row
        const bf16_t* gsrc = (seg < 2)
            ? (xbf + ((size_t)row * SEQ + (size_t)s) * HID + seg * 512)
            : (h_in + (size_t)row * HID + (seg - 2) * 512);
        const unsigned lbase = (unsigned)(row * LDA + seg * 512) * 2u;
        #pragma unroll 8
        for (int j = 0; j < 64; ++j) {       // 64 x 16B = 1KB per thread
            async_copy_b128(lbase + (unsigned)j * 16u,
                            (unsigned long long)(uintptr_t)gsrc + (unsigned long long)j * 16ull);
        }
        wait_async0();
    }
    __syncthreads();

    // ---- WMMA main loop ----
    const int wave  = tid >> 5;          // 0..7
    const int lane  = tid & 31;
    const int g     = wave >> 1;         // gate 0..3
    const int nhalf = wave & 1;          // n offset nhalf*32
    const int lrow  = lane & 15;
    const int khalf = lane >> 4;         // 0/1: which K-half this lane holds

    v8f acc[4][2];
    #pragma unroll
    for (int m = 0; m < 4; ++m)
        #pragma unroll
        for (int nn = 0; nn < 2; ++nn)
            acc[m][nn] = (v8f){0.f,0.f,0.f,0.f,0.f,0.f,0.f,0.f};

    // A-frag: LDS row = m*16 + lrow. B-frag: weight row = g*HID + n0 + nhalf*32 + nn*16 + lrow
    // (K contiguous in memory -> identical fragment addressing pattern).
    const bf16_t* arow[4];
    #pragma unroll
    for (int m = 0; m < 4; ++m)
        arow[m] = A + (size_t)(m * 16 + lrow) * LDA;
    const bf16_t* wrow[2];
    #pragma unroll
    for (int nn = 0; nn < 2; ++nn)
        wrow[nn] = Wcat + (size_t)(g * HID + n0 + nhalf * 32 + nn * 16 + lrow) * K2;

    for (int k0 = 0; k0 < K2; k0 += 32) {
        const int off0 = k0 + khalf * 8;       // K = khalf*8 + {0..7}
        const int off1 = k0 + 16 + khalf * 8;  // K = 16 + khalf*8 + {0..7}

        v16bf afr[4], bfr[2];
        #pragma unroll
        for (int m = 0; m < 4; ++m)
            afr[m] = concat8(*(const v8bf*)(arow[m] + off0),
                             *(const v8bf*)(arow[m] + off1));
        #pragma unroll
        for (int nn = 0; nn < 2; ++nn)
            bfr[nn] = concat8(*(const v8bf*)(wrow[nn] + off0),
                              *(const v8bf*)(wrow[nn] + off1));

        #pragma unroll
        for (int m = 0; m < 4; ++m)
            #pragma unroll
            for (int nn = 0; nn < 2; ++nn)
                acc[m][nn] = __builtin_amdgcn_wmma_f32_16x16x32_bf16(
                    /*neg_a=*/false, afr[m],
                    /*neg_b=*/false, bfr[nn],
                    /*c_mod=*/(short)0, acc[m][nn],
                    /*reuse_a=*/false, /*reuse_b=*/false);
    }
    __syncthreads();   // all A reads complete before LDS reuse

    // ---- spill accumulators to LDS gate buffer Gb[g][batch_row][k_in_tile] ----
    // C/D layout: VGPR r -> M = r + 8*khalf, N = lrow.
    #pragma unroll
    for (int m = 0; m < 4; ++m) {
        #pragma unroll
        for (int nn = 0; nn < 2; ++nn) {
            const int col = nhalf * 32 + nn * 16 + lrow;
            #pragma unroll
            for (int r = 0; r < 8; ++r) {
                const int row = m * 16 + r + 8 * khalf;
                Gb[((g * 64 + row) << 6) + col] = acc[m][nn][r];
            }
        }
    }
    __syncthreads();

    // ---- fused peephole-LSTM cell update epilogue ----
    {
        const int b   = tid >> 2;            // batch row 0..63
        const int kk0 = (tid & 3) * 16;      // 16 k's per thread
        #pragma unroll 4
        for (int j = 0; j < 16; ++j) {
            const int kt = kk0 + j;
            const int kg = n0 + kt;
            const float pre_i = Gb[((0 * 64 + b) << 6) + kt] + bias[0 * HID + kg];
            const float pre_f = Gb[((1 * 64 + b) << 6) + kt] + bias[1 * HID + kg];
            const float pre_c = Gb[((2 * 64 + b) << 6) + kt] + bias[2 * HID + kg];
            const float pre_o = Gb[((3 * 64 + b) << 6) + kt] + bias[3 * HID + kg];
            const float c_old = cbuf[b * HID + kg];
            const float ig = sigmoidf_(pre_i + peep[0 * HID + kg] * c_old);
            const float fg = sigmoidf_(pre_f + peep[1 * HID + kg] * c_old);
            const float cn = fg * c_old + ig + tanhf(pre_c);   // note: +ig per reference
            const float og = sigmoidf_(pre_o + peep[2 * HID + kg] * cn);
            const float hn = og * tanhf(cn);
            cbuf[b * HID + kg]  = cn;
            h_out[b * HID + kg] = (bf16_t)hn;      // next step's GEMM operand
            out[((size_t)b * SEQ + (size_t)s) * HID + kg] = hn;
        }
    }
}

// ---------------- host launcher ----------------
extern "C" void kernel_launch(void* const* d_in, const int* in_sizes, int n_in,
                              void* d_out, int out_size, void* d_ws, size_t ws_size,
                              hipStream_t stream) {
    (void)in_sizes; (void)n_in; (void)out_size; (void)ws_size;
    const float* x     = (const float*)d_in[0];
    const float* h0    = (const float*)d_in[1];
    const float* c0    = (const float*)d_in[2];
    const float* Wx    = (const float*)d_in[3];
    const float* bx    = (const float*)d_in[4];
    const float* Wh    = (const float*)d_in[5];
    const float* bh    = (const float*)d_in[6];
    const float* peep  = (const float*)d_in[7];
    const float* bgate = (const float*)d_in[8];
    float* out = (float*)d_out;

    // workspace layout (~50.5 MB total)
    char*   ws    = (char*)d_ws;
    bf16_t* Wcat  = (bf16_t*)ws;                                      // 16 MiB
    bf16_t* xbf   = (bf16_t*)(ws + (size_t)GATES * HID * K2 * 2);     // 32 MiB
    float*  bias  = (float*)((char*)xbf + (size_t)BATCH * SEQ * HID * 2);  // 16 KiB
    bf16_t* hbf0  = (bf16_t*)(bias + GATES * HID);                    // 128 KiB
    bf16_t* hbf1  = hbf0 + BATCH * HID;                               // 128 KiB
    float*  cbuf  = (float*)(hbf1 + BATCH * HID);                     // 256 KiB

    {
        const size_t total = (size_t)GATES * HID * K2;
        convlstm_pack_weights<<<(unsigned)((total + 255) / 256), 256, 0, stream>>>(Wx, Wh, Wcat);
    }
    {
        const size_t totalx = (size_t)BATCH * SEQ * HID;              // 16,777,216
        convlstm_convert_x<<<(unsigned)(totalx / 8 / 256), 256, 0, stream>>>(x, xbf);
    }
    convlstm_pack_bias<<<(GATES * HID + 255) / 256, 256, 0, stream>>>(bx, bh, bgate, bias);
    convlstm_init_state<<<(BATCH * HID + 255) / 256, 256, 0, stream>>>(h0, c0, hbf0, cbuf);

    const size_t ldsBytes = (size_t)BATCH * LDA * sizeof(bf16_t);  // 263,168 B (CDNA5 WGP LDS)
    (void)hipFuncSetAttribute((const void*)convlstm_step,
                              hipFuncAttributeMaxDynamicSharedMemorySize,
                              (int)ldsBytes);

    for (int s = 0; s < SEQ; ++s) {
        const bf16_t* hin  = (s & 1) ? hbf1 : hbf0;
        bf16_t*       hout = (s & 1) ? hbf0 : hbf1;
        convlstm_step<<<HID / NTILE, 256, ldsBytes, stream>>>(
            xbf, Wcat, bias, peep, hin, hout, cbuf, out, s);
    }
}
